// TimeAwareCondConv2d_60885456388308
// MI455X (gfx1250) — matrix operand order repaired and checked by reading, not compiled, not run on hardware
//
#include <hip/hip_runtime.h>
#include <math.h>

// ---------------- types for WMMA ----------------
typedef __bf16 bf16_t;
typedef bf16_t v16bf __attribute__((ext_vector_type(16)));
typedef bf16_t v8bf  __attribute__((ext_vector_type(8)));
typedef float  v8f   __attribute__((ext_vector_type(8)));

// Problem constants
#define BB 32
#define CC 128
#define HH 64
#define WW 64
#define OC 128
#define EE 8
#define TT 256
#define HID 64
#define UP 256
#define RR 32
#define GH 8
#define KDIM (9*CC)          // 1152
#define PIX (HH*WW)          // 4096

// LDS row layout for the conv kernel: 3 rows x 66 pixels (1 guard each side) x 128 ch
#define LROW (66*128)        // ushorts per padded row = 8448

__device__ __forceinline__ unsigned short f2bf(float f) {
    unsigned int u = __float_as_uint(f);
    unsigned int r = u + 0x7FFFu + ((u >> 16) & 1u);   // round-to-nearest-even
    return (unsigned short)(r >> 16);
}
__device__ __forceinline__ float sigm(float z) { return 1.0f / (1.0f + expf(-z)); }

// ---------------------------------------------------------------------------
// Kernel 1: per-(b,c) mean / population-std over 64x64 spatial
// ---------------------------------------------------------------------------
__global__ __launch_bounds__(256)
void stats_kernel(const float* __restrict__ x, float* __restrict__ mu, float* __restrict__ sd) {
    __shared__ float ssum[256], ssq[256];
    const int bc = blockIdx.x;                 // b*128 + c
    const float* p = x + (size_t)bc * PIX;
    float s = 0.f, q = 0.f;
    for (int i = threadIdx.x; i < PIX; i += 256) { float v = p[i]; s += v; q += v * v; }
    ssum[threadIdx.x] = s; ssq[threadIdx.x] = q;
    __syncthreads();
    for (int off = 128; off > 0; off >>= 1) {
        if ((int)threadIdx.x < off) {
            ssum[threadIdx.x] += ssum[threadIdx.x + off];
            ssq[threadIdx.x]  += ssq[threadIdx.x + off];
        }
        __syncthreads();
    }
    if (threadIdx.x == 0) {
        float m   = ssum[0] * (1.0f / PIX);
        float var = ssq[0] * (1.0f / PIX) - m * m;
        mu[bc] = m;
        sd[bc] = sqrtf(fmaxf(var, 0.0f)) + 1e-6f;   // +EPS folded in
    }
}

// ---------------------------------------------------------------------------
// Kernel 2: NCHW f32 -> NHWC bf16 transpose  xT[b][y][x][c]
// grid = B*64 blocks, each handles 64 pixels x 128 channels
// ---------------------------------------------------------------------------
__global__ __launch_bounds__(256)
void transpose_bf16_kernel(const float* __restrict__ x, unsigned short* __restrict__ xT) {
    __shared__ unsigned short tile[64 * 128];    // 16 KB
    const int blk = blockIdx.x;
    const int b = blk >> 6;
    const int pixbase = (blk & 63) * 64;
    const int t = threadIdx.x;
    const int p = t & 63;
    const int cq = t >> 6;                       // 0..3
    const float* xb = x + (size_t)b * (CC * PIX);
    #pragma unroll 4
    for (int i = 0; i < 32; ++i) {
        int c = cq * 32 + i;
        float v = xb[(size_t)c * PIX + pixbase + p];
        tile[p * 128 + c] = f2bf(v);
    }
    __syncthreads();
    const unsigned int* src = (const unsigned int*)tile;
    unsigned int* dst = (unsigned int*)(xT + ((size_t)b * PIX + pixbase) * CC);
    #pragma unroll
    for (int i = 0; i < 16; ++i) dst[t + 256 * i] = src[t + 256 * i];
}

// ---------------------------------------------------------------------------
// Kernel 3: router MLP + softmax -> rw[B,E]; one block per batch element
// ---------------------------------------------------------------------------
__global__ __launch_bounds__(256)
void router_kernel(const float* __restrict__ mu, const float* __restrict__ sd,
                   const float* __restrict__ time_emb,
                   const float* __restrict__ W_q, const float* __restrict__ W_k,
                   const float* __restrict__ W_v,
                   const float* __restrict__ W_g, const float* __restrict__ b_g,
                   const float* __restrict__ W_a1, const float* __restrict__ b_a1,
                   const float* __restrict__ W_a2,
                   const float* __restrict__ W_b1, const float* __restrict__ b_b1,
                   const float* __restrict__ W_b2,
                   const float* __restrict__ W_d, const float* __restrict__ W_out,
                   float* __restrict__ rw)
{
    __shared__ float stats[2 * CC], xh[HID], gate[UP], a1s[GH], b1s[GH],
                     tv[RR], mbuf[UP], xm[HID], logits[EE];
    const int b = blockIdx.x;
    const int t = threadIdx.x;

    stats[t] = (t < CC) ? mu[b * CC + t] : sd[b * CC + (t - CC)];
    __syncthreads();

    if (t < HID) {
        float q = 0.f, k = 0.f, v = 0.f;
        const float* te = time_emb + b * TT;
        for (int j = 0; j < TT; ++j) {
            q += te[j] * W_q[t * TT + j];
            float s = stats[j];
            k += s * W_k[t * (2 * CC) + j];
            v += s * W_v[t * (2 * CC) + j];
        }
        xh[t] = v * sigm(q * k);
    }
    __syncthreads();

    { // gate (all 256 threads)
        float g = b_g[t];
        for (int h = 0; h < HID; ++h) g += xh[h] * W_g[t * HID + h];
        gate[t] = g;
    }
    if (t < 16) { // a1 / b1  (silu of small MLP)
        const float* Wm = (t < GH) ? W_a1 : W_b1;
        const float* bm = (t < GH) ? b_a1 : b_b1;
        int g = t & (GH - 1);
        float a = bm[g];
        for (int h = 0; h < HID; ++h) a += xh[h] * Wm[g * HID + h];
        a = a * sigm(a);
        if (t < GH) a1s[g] = a; else b1s[g] = a;
    }
    __syncthreads();

    if (t < RR) { // tv[r] = sum_h xh[h] * A[h][r],  A from W_a2 @ a1
        float acc = 0.f;
        for (int h = 0; h < HID; ++h) {
            const float* w = W_a2 + (size_t)(h * RR + t) * GH;
            float av = 0.f;
            #pragma unroll
            for (int g = 0; g < GH; ++g) av += a1s[g] * w[g];
            acc += xh[h] * av;
        }
        tv[t] = acc;
    }
    __syncthreads();

    { // dyn + silu(gate)*dyn
        float d = 0.f;
        for (int r = 0; r < RR; ++r) {
            const float* w = W_b2 + (size_t)(r * UP + t) * GH;
            float bv = 0.f;
            #pragma unroll
            for (int g = 0; g < GH; ++g) bv += b1s[g] * w[g];
            d += tv[r] * bv;
        }
        float ga = gate[t];
        mbuf[t] = (ga * sigm(ga)) * d;
    }
    __syncthreads();

    if (t < HID) {
        float acc = xh[t];
        for (int u = 0; u < UP; ++u) acc += mbuf[u] * W_d[t * UP + u];
        xm[t] = acc;
    }
    __syncthreads();

    if (t < EE) {
        float l = 0.f;
        for (int h = 0; h < HID; ++h) l += xm[h] * W_out[t * HID + h];
        logits[t] = l;
    }
    __syncthreads();

    if (t == 0) {
        float mx = logits[0];
        for (int e = 1; e < EE; ++e) mx = fmaxf(mx, logits[e]);
        float s = 0.f, ex[EE];
        for (int e = 0; e < EE; ++e) { ex[e] = expf(logits[e] - mx); s += ex[e]; }
        float inv = 1.0f / s;
        for (int e = 0; e < EE; ++e) rw[b * EE + e] = ex[e] * inv;
    }
}

// ---------------------------------------------------------------------------
// Kernel 4: expert mix -> bf16 weights in GEMM-friendly layout [b][oc][r][c]
// grid = B*OC blocks
// ---------------------------------------------------------------------------
__global__ __launch_bounds__(256)
void mix_kernel(const float* __restrict__ expert_w, const float* __restrict__ rw,
                unsigned short* __restrict__ wbf)
{
    const int boc = blockIdx.x;                  // b*128 + oc
    const int b = boc >> 7;
    const int oc = boc & 127;
    float w[EE];
    #pragma unroll
    for (int e = 0; e < EE; ++e) w[e] = rw[b * EE + e];
    const size_t eb = (size_t)oc * (CC * 9);
    unsigned short* dst = wbf + (size_t)boc * KDIM;
    for (int i = threadIdx.x; i < KDIM; i += 256) {
        int r = i >> 7;                          // tap index (ky*3+kx)
        int c = i & 127;                         // channel
        size_t src = eb + (size_t)c * 9 + r;     // expert_w[e][oc][c][ky][kx]
        float acc = 0.f;
        #pragma unroll
        for (int e = 0; e < EE; ++e) acc += w[e] * expert_w[(size_t)e * (OC * CC * 9) + src];
        dst[i] = f2bf(acc);
    }
}

// ---------------------------------------------------------------------------
// Kernel 5: implicit-GEMM conv via V_WMMA_F32_16X16X32_BF16
// One block per (b, y) output row; 8 waves = 4 oc-tiles x 2 x-halves.
// Input rows y-1, y, y+1 staged once in LDS with zero guard pixels, so the
// inner loop is branch-free: 2 global A-frag loads + 2 LDS B-frag loads
// per 4 WMMAs.
// ---------------------------------------------------------------------------
__device__ __forceinline__ v16bf load_afrag(const unsigned short* p) {
    v8bf lo = *(const v8bf*)(p);
    v8bf hi = *(const v8bf*)(p + 8);
    return __builtin_shufflevector(lo, hi, 0,1,2,3,4,5,6,7,8,9,10,11,12,13,14,15);
}

__global__ __launch_bounds__(256)
void conv_wmma_kernel(const unsigned short* __restrict__ xT,   // [B][64][64][128] bf16
                      const unsigned short* __restrict__ wbf,  // [B][128][9][128] bf16
                      float* __restrict__ out)                 // [B][128][64][64] f32
{
    __shared__ unsigned short smem[3 * LROW];    // 3 padded rows = 49.5 KB

    const int blk  = blockIdx.x;                 // B*64 blocks
    const int b    = blk >> 6;
    const int y    = blk & 63;
    const int tid  = threadIdx.x;
    const int lane = tid & 31;
    const int wave = tid >> 5;
    const int mt   = wave >> 1;                  // oc tile of 32
    const int xb   = (wave & 1) << 5;            // x half: 0 / 32
    const int m0   = mt << 5;

    const int lhalf = lane >> 4;                 // 0 / 1 (wave half)
    const int l16   = lane & 15;

    // ---- stage input rows y-1, y, y+1 into LDS (bf16, guard pixel each side) ----
    {
        // zero the guard pixels (128 ushorts = 64 uints per guard, 2 guards x 3 rows)
        unsigned int* sm32 = (unsigned int*)smem;
        if (tid < 64) {
            #pragma unroll
            for (int r3 = 0; r3 < 3; ++r3) {
                sm32[(r3 * LROW) / 2 + tid]            = 0u;   // pixel 0
                sm32[(r3 * LROW + 65 * 128) / 2 + tid] = 0u;   // pixel 65
            }
        }
        const uint4* gsrc_base = (const uint4*)(xT + (size_t)b * (PIX * CC));
        #pragma unroll
        for (int r3 = 0; r3 < 3; ++r3) {
            const int yy = y + r3 - 1;
            uint4* ldst = (uint4*)(smem + r3 * LROW + 128);     // skip guard pixel
            if ((unsigned)yy < (unsigned)HH) {
                const uint4* gsrc = gsrc_base + (size_t)yy * (WW * CC / 8); // 1024 uint4 per row
                #pragma unroll
                for (int i = 0; i < 4; ++i) ldst[tid + 256 * i] = gsrc[tid + 256 * i];
            } else {
                const uint4 z = {0u, 0u, 0u, 0u};
                #pragma unroll
                for (int i = 0; i < 4; ++i) ldst[tid + 256 * i] = z;
            }
        }
    }
    __syncthreads();

    v8f acc00 = {}; v8f acc01 = {}; v8f acc10 = {}; v8f acc11 = {};

    // A-matrix row pointers (global, L2-resident weights)
    const unsigned short* wb    = wbf + (size_t)(b * OC) * KDIM;
    const unsigned short* arow0 = wb + (size_t)(m0 + l16) * KDIM;
    const unsigned short* arow1 = wb + (size_t)(m0 + 16 + l16) * KDIM;
    const int akoff = lhalf * 16;

    // B fragment LDS base for this lane: pixel index = x + 1 (guard offset)
    const int bbase = (xb + l16 + 1) * 128 + (lhalf << 4);

    for (int r = 0; r < 9; ++r) {
        const int dy = r / 3;                    // 0..2 -> LDS row index directly
        const int dx = r % 3 - 1;                // -1..1
        const unsigned short* brow = smem + dy * LROW + bbase + dx * 128;
        const unsigned short* a0p  = arow0 + r * CC + akoff;
        const unsigned short* a1p  = arow1 + r * CC + akoff;
        #pragma unroll
        for (int cb = 0; cb < CC; cb += 32) {
            v16bf a0  = load_afrag(a0p + cb);
            v16bf a1  = load_afrag(a1p + cb);
            v16bf bf0 = *(const v16bf*)(brow + cb);             // n-sub 0
            v16bf bf1 = *(const v16bf*)(brow + 16 * 128 + cb);  // n-sub 1
            acc00 = __builtin_amdgcn_wmma_f32_16x16x32_bf16(false, a0, false, bf0, (short)0, acc00, false, false);
            acc01 = __builtin_amdgcn_wmma_f32_16x16x32_bf16(false, a0, false, bf1, (short)0, acc01, false, false);
            acc10 = __builtin_amdgcn_wmma_f32_16x16x32_bf16(false, a1, false, bf0, (short)0, acc10, false, false);
            acc11 = __builtin_amdgcn_wmma_f32_16x16x32_bf16(false, a1, false, bf1, (short)0, acc11, false, false);
        }
    }

    // D layout: VGPR reg -> M = reg + 8*lhalf ; N = l16
    float* ob = out + (size_t)b * (OC * PIX);
    const int xs0 = xb + l16;
    #pragma unroll
    for (int reg = 0; reg < 8; ++reg) {
        int oc0 = m0 + reg + lhalf * 8;
        int oc1 = oc0 + 16;
        float* o0 = ob + ((size_t)oc0 * HH + y) * WW;
        float* o1 = ob + ((size_t)oc1 * HH + y) * WW;
        o0[xs0]      = acc00[reg];
        o0[xs0 + 16] = acc01[reg];
        o1[xs0]      = acc10[reg];
        o1[xs0 + 16] = acc11[reg];
    }
}

// ---------------------------------------------------------------------------
extern "C" void kernel_launch(void* const* d_in, const int* in_sizes, int n_in,
                              void* d_out, int out_size, void* d_ws, size_t ws_size,
                              hipStream_t stream)
{
    const float* x        = (const float*)d_in[0];
    const float* time_emb = (const float*)d_in[1];
    const float* expert_w = (const float*)d_in[2];
    const float* W_q   = (const float*)d_in[3];
    const float* W_k   = (const float*)d_in[4];
    const float* W_v   = (const float*)d_in[5];
    const float* W_g   = (const float*)d_in[6];
    const float* b_g   = (const float*)d_in[7];
    const float* W_a1  = (const float*)d_in[8];
    const float* b_a1  = (const float*)d_in[9];
    const float* W_a2  = (const float*)d_in[10];
    const float* W_b1  = (const float*)d_in[11];
    const float* b_b1  = (const float*)d_in[12];
    const float* W_b2  = (const float*)d_in[13];
    const float* W_d   = (const float*)d_in[14];
    const float* W_out = (const float*)d_in[15];

    char* ws = (char*)d_ws;
    float* mu           = (float*)(ws + 0);                    // 16 KB
    float* sd           = (float*)(ws + 16384);                // 16 KB
    float* rw           = (float*)(ws + 32768);                // 1 KB
    unsigned short* wbf = (unsigned short*)(ws + 65536);       // 9,437,184 B
    unsigned short* xT  = (unsigned short*)(ws + 9502720);     // 33,554,432 B (total ~43 MB)

    stats_kernel         <<<BB * CC, 256, 0, stream>>>(x, mu, sd);
    transpose_bf16_kernel<<<BB * 64, 256, 0, stream>>>(x, xT);
    router_kernel        <<<BB,      256, 0, stream>>>(mu, sd, time_emb,
                                                       W_q, W_k, W_v, W_g, b_g,
                                                       W_a1, b_a1, W_a2,
                                                       W_b1, b_b1, W_b2,
                                                       W_d, W_out, rw);
    mix_kernel           <<<BB * OC, 256, 0, stream>>>(expert_w, rw, wbf);
    conv_wmma_kernel     <<<BB * 64, 256, 0, stream>>>(xT, wbf, (float*)d_out);
}
